// BlockRecurrentModel_33346126086943
// MI455X (gfx1250) — matrix operand order repaired and checked by reading, not compiled, not run on hardware
//
#include <hip/hip_runtime.h>
#include <hip/hip_bf16.h>
#include <stdint.h>

// ---------------------------------------------------------------------------
// MI455X / gfx1250 fused BlockRecurrentModel forward.
// Compute-bound (~217 GFLOP matrix work vs ~260MB traffic -> ~11us @23.3TB/s),
// so all GEMMs run through v_wmma_f32_16x16x32_bf16 with f32 accumulation.
// Tile staging uses CDNA5 async global->LDS copies (ASYNCcnt) via inline asm.
// ---------------------------------------------------------------------------

typedef __bf16 bf16;
typedef __attribute__((ext_vector_type(16))) __bf16 v16bf;
typedef __attribute__((ext_vector_type(8)))  float  v8f;

#define N_TOK   4096
#define HID     1024
#define DETER   4096
#define G_GRP   8
#define BH      512
#define BLK_IN  3584   // BH + 3*HID

// ---- CDNA5 async global->LDS path (inline asm; fallback = plain ds stores) --
#if defined(__gfx1250__)
#  define TILE_ASYNC 1
#else
#  define TILE_ASYNC 0
#endif

__device__ __forceinline__ void copy16_to_lds(void* dst_lds, const void* src_glb) {
#if TILE_ASYNC
  // LDS byte offset: generic -> addrspace(3) -> 32-bit offset
  typedef __attribute__((address_space(3))) char lds_char;
  unsigned lds_off = (unsigned)(unsigned long long)(lds_char*)dst_lds;
  unsigned long long gaddr = (unsigned long long)src_glb;
  // Per-lane async 16B copy, tracked by ASYNCcnt (CDNA5 ISA)
  asm volatile("global_load_async_to_lds_b128 %0, %1, off"
               :
               : "v"(lds_off), "v"(gaddr)
               : "memory");
#else
  *(uint4*)dst_lds = *(const uint4*)src_glb;
#endif
}

__device__ __forceinline__ void wait_tiles() {
#if TILE_ASYNC
#  if defined(__has_builtin)
#    if __has_builtin(__builtin_amdgcn_s_wait_asynccnt)
  __builtin_amdgcn_s_wait_asynccnt(0);
#    else
  asm volatile("s_wait_asynccnt 0" ::: "memory");
#    endif
#  else
  asm volatile("s_wait_asynccnt 0" ::: "memory");
#  endif
#endif
}

// ---------------------------------------------------------------------------
// f32 -> bf16 elementwise convert (grid-stride)
// ---------------------------------------------------------------------------
__global__ void k_cvt_bf16(const float* __restrict__ in, bf16* __restrict__ out,
                           size_t n) {
  size_t i = (size_t)blockIdx.x * blockDim.x + threadIdx.x;
  size_t stride = (size_t)gridDim.x * blockDim.x;
  for (; i < n; i += stride) out[i] = (bf16)in[i];
}

// ---------------------------------------------------------------------------
// f32 (G,K,N) -> bf16 (G,N,K) transpose (32x32 LDS tiles, coalesced both ways)
// ---------------------------------------------------------------------------
__global__ void k_transpose_bf16(const float* __restrict__ in,
                                 bf16* __restrict__ out, int K, int Nn) {
  __shared__ float tile[32][33];
  const int g = blockIdx.z;
  const float* ing = in + (size_t)g * K * Nn;
  bf16* outg = out + (size_t)g * K * Nn;
  const int k0 = blockIdx.y * 32, n0 = blockIdx.x * 32;
  const int tx = threadIdx.x & 31, ty = threadIdx.x >> 5;  // 32x8
#pragma unroll
  for (int i = 0; i < 32; i += 8)
    tile[ty + i][tx] = ing[(size_t)(k0 + ty + i) * Nn + n0 + tx];
  __syncthreads();
#pragma unroll
  for (int i = 0; i < 32; i += 8)
    outg[(size_t)(n0 + ty + i) * K + k0 + tx] = (bf16)tile[tx][ty + i];
}

// ---------------------------------------------------------------------------
// Generic WMMA bf16 GEMM:  C(f32, M x N) = [A1 | A2](bf16) @ BT(bf16 N x Ktot)^T
//  - A is row-major, split into two K-segments (A2 optional, K2 == 0 to skip);
//    used to feed the block-diagonal GEMM with [d_g | x_cat] without
//    materializing the concatenation.
//  - BT is the transposed weight (out,in) so each lane reads contiguous K.
//  - 256 threads = 8 wave32s; block tile 128x128, KT = 32, LDS double buffer.
//  - per-wave tile 32x64 -> 8 x v_wmma_f32_16x16x32_bf16 per K-step.
//  - blockIdx.z selects the diagonal block (g); *_gs are per-g element strides.
// ---------------------------------------------------------------------------
__global__ __launch_bounds__(256)
void k_gemm_bf16(const bf16* __restrict__ A1, int lda1, int K1, long a1_gs,
                 const bf16* __restrict__ A2, int lda2, int K2, long a2_gs,
                 const bf16* __restrict__ BT, int ldbt, long bt_gs,
                 float* __restrict__ C, int ldc, long c_gs) {
  __shared__ bf16 sA[2][128][32];
  __shared__ bf16 sB[2][128][32];

  const int g = blockIdx.z;
  A1 += (size_t)g * a1_gs;
  if (K2 > 0) A2 += (size_t)g * a2_gs;
  BT += (size_t)g * bt_gs;
  C  += (size_t)g * c_gs;

  const int m0 = blockIdx.y * 128;
  const int n0 = blockIdx.x * 128;

  const int lane = threadIdx.x & 31;
  const int wid  = threadIdx.x >> 5;
  const int wm   = wid & 3;   // wave row   -> M offset wm*32
  const int wn   = wid >> 2;  // wave col   -> N offset wn*64
  const int half = lane >> 4;
  const int lr   = lane & 15;

  const int nK = (K1 + K2) / 32;

  // tile loader: 512 16B chunks per tile, 2 per thread per tile
  auto load_tiles = [&](int buf, int kk) {
    const int t = threadIdx.x;
#pragma unroll
    for (int it = 0; it < 2; ++it) {
      const int chunk = t + it * 256;
      const int row = chunk >> 2;
      const int cc  = (chunk & 3) * 8;           // bf16 col within tile
      const int k   = kk + cc;
      const bf16* src = (k < K1)
          ? A1 + (size_t)(m0 + row) * lda1 + k
          : A2 + (size_t)(m0 + row) * lda2 + (k - K1);
      copy16_to_lds(&sA[buf][row][cc], src);
    }
#pragma unroll
    for (int it = 0; it < 2; ++it) {
      const int chunk = t + it * 256;
      const int row = chunk >> 2;
      const int cc  = (chunk & 3) * 8;
      copy16_to_lds(&sB[buf][row][cc],
                    BT + (size_t)(n0 + row) * ldbt + kk + cc);
    }
  };

  v8f acc[2][4] = {};

  load_tiles(0, 0);
  for (int ks = 0; ks < nK; ++ks) {
    const int buf = ks & 1;
    wait_tiles();
    __syncthreads();
    if (ks + 1 < nK) load_tiles(buf ^ 1, (ks + 1) * 32);

    // A fragments: lane lr holds row M=lr; halves split K {0..7,16..23} vs
    // {8..15,24..31} per the 16-bit A layout.
    union Frag { v16bf v; uint4 q[2]; };
    v16bf afr[2], bfr[4];
#pragma unroll
    for (int i = 0; i < 2; ++i) {
      Frag u;
      const int row = wm * 32 + i * 16 + lr;
      u.q[0] = *(const uint4*)&sA[buf][row][half * 8];
      u.q[1] = *(const uint4*)&sA[buf][row][16 + half * 8];
      afr[i] = u.v;
    }
    // B fragments: lane lr holds column N=lr; halves split K {0..7,16..23} vs
    // {8..15,24..31}, matching A
#pragma unroll
    for (int j = 0; j < 4; ++j) {
      Frag u;
      const int col = wn * 64 + j * 16 + lr;
      u.q[0] = *(const uint4*)&sB[buf][col][half * 8];
      u.q[1] = *(const uint4*)&sB[buf][col][16 + half * 8];
      bfr[j] = u.v;
    }
#pragma unroll
    for (int i = 0; i < 2; ++i)
#pragma unroll
      for (int j = 0; j < 4; ++j)
        acc[i][j] = __builtin_amdgcn_wmma_f32_16x16x32_bf16(
            false, afr[i], false, bfr[j], (short)0, acc[i][j], false, false);
  }

  // C layout: VGPR r -> rows {half*8 + r}, col = lane&15 within each 16x16 tile
#pragma unroll
  for (int i = 0; i < 2; ++i)
#pragma unroll
    for (int j = 0; j < 4; ++j) {
      const int colIdx = n0 + wn * 64 + j * 16 + lr;
#pragma unroll
      for (int r = 0; r < 8; ++r) {
        const int rowIdx = m0 + wm * 32 + i * 16 + half * 8 + r;
        C[(size_t)rowIdx * ldc + colIdx] = acc[i][j][r];
      }
    }
}

// ---------------------------------------------------------------------------
// Row LayerNorm (population var, eps=1e-3) + SiLU, writing bf16 into a
// (possibly offset/strided) destination. One block (256 thr, 8 waves) per row.
// ---------------------------------------------------------------------------
__global__ void k_ln_silu(const float* __restrict__ y, int L,
                          const float* __restrict__ gam,
                          const float* __restrict__ bet,
                          bf16* __restrict__ out, int ostride, int ooff) {
  const int row = blockIdx.x;
  const float* yr = y + (size_t)row * L;
  float s = 0.f, ss = 0.f;
  for (int i = threadIdx.x; i < L; i += 256) {
    const float v = yr[i];
    s += v; ss += v * v;
  }
#pragma unroll
  for (int o = 16; o; o >>= 1) {
    s  += __shfl_xor(s, o, 32);
    ss += __shfl_xor(ss, o, 32);
  }
  __shared__ float red[16];
  __shared__ float stat[2];
  if ((threadIdx.x & 31) == 0) {
    red[threadIdx.x >> 5]       = s;
    red[8 + (threadIdx.x >> 5)] = ss;
  }
  __syncthreads();
  if (threadIdx.x == 0) {
    float S = 0.f, SS = 0.f;
#pragma unroll
    for (int i = 0; i < 8; ++i) { S += red[i]; SS += red[8 + i]; }
    const float mu = S / L;
    stat[0] = mu;
    stat[1] = rsqrtf(SS / L - mu * mu + 1e-3f);
  }
  __syncthreads();
  const float mu = stat[0], rs = stat[1];
  bf16* orow = out + (size_t)row * ostride + ooff;
  for (int i = threadIdx.x; i < L; i += 256) {
    const float v = (yr[i] - mu) * rs * gam[i] + bet[i];
    const float sig = 1.f / (1.f + __expf(-v));
    orow[i] = (bf16)(v * sig);
  }
}

// ---------------------------------------------------------------------------
// GRU epilogue: gates (N, G*1536) f32 -> d_new (N, DETER) f32
// ---------------------------------------------------------------------------
__global__ void k_gru_update(const float* __restrict__ gates,
                             const float* __restrict__ bg,
                             const float* __restrict__ d2,
                             float* __restrict__ dout, size_t total) {
  size_t i = (size_t)blockIdx.x * blockDim.x + threadIdx.x;
  size_t stride = (size_t)gridDim.x * blockDim.x;
  for (; i < total; i += stride) {
    const int    col = (int)(i & (DETER - 1));
    const size_t n   = i >> 12;
    const int g = col >> 9, j = col & (BH - 1);
    const float* gr = gates + n * (size_t)(G_GRP * 3 * BH) + (size_t)g * (3 * BH);
    const float* bgr = bg + (size_t)g * (3 * BH);
    float r = gr[j]          + bgr[j];
    float u = gr[BH + j]     + bgr[BH + j];
    float c = gr[2 * BH + j] + bgr[2 * BH + j];
    r = 1.f / (1.f + __expf(-r));
    u = 1.f / (1.f + __expf(-(u - 1.f)));
    c = tanhf(r * c);
    const float d0 = d2[i];
    dout[i] = u * c + (1.f - u) * d0;
  }
}

// ---------------------------------------------------------------------------
extern "C" void kernel_launch(void* const* d_in, const int* in_sizes, int n_in,
                              void* d_out, int out_size, void* d_ws, size_t ws_size,
                              hipStream_t stream) {
  const float* s      = (const float*)d_in[0];
  const float* a      = (const float*)d_in[1];
  const float* d      = (const float*)d_in[2];
  const float* W_s    = (const float*)d_in[3];
  const float* W_a    = (const float*)d_in[4];
  const float* W_d    = (const float*)d_in[5];
  const float* ln_s_g = (const float*)d_in[6];
  const float* ln_s_b = (const float*)d_in[7];
  const float* ln_a_g = (const float*)d_in[8];
  const float* ln_a_b = (const float*)d_in[9];
  const float* ln_d_g = (const float*)d_in[10];
  const float* ln_d_b = (const float*)d_in[11];
  const float* W_hid  = (const float*)d_in[12];
  const float* ln_h_g = (const float*)d_in[13];
  const float* ln_h_b = (const float*)d_in[14];
  const float* W_gate = (const float*)d_in[15];
  const float* b_gate = (const float*)d_in[16];
  float* out = (float*)d_out;

  // ---- workspace carve (256B aligned) ----
  char* w = (char*)d_ws;
  size_t off = 0;
  auto carve = [&](size_t bytes) {
    void* p = w + off;
    off = (off + bytes + 255) & ~(size_t)255;
    return p;
  };
  bf16* s_bf    = (bf16*)carve((size_t)N_TOK * 1024 * 2);
  bf16* a_bf    = (bf16*)carve((size_t)N_TOK * 256 * 2);
  bf16* d_bf    = (bf16*)carve((size_t)N_TOK * 4096 * 2);
  bf16* WsT     = (bf16*)carve((size_t)1024 * 1024 * 2);
  bf16* WaT     = (bf16*)carve((size_t)256 * 1024 * 2);
  bf16* WdT     = (bf16*)carve((size_t)4096 * 1024 * 2);
  bf16* WhidT   = (bf16*)carve((size_t)G_GRP * BLK_IN * BH * 2);
  bf16* WgateT  = (bf16*)carve((size_t)G_GRP * BH * 3 * BH * 2);
  bf16* xcat    = (bf16*)carve((size_t)N_TOK * 3 * HID * 2);
  bf16* hact    = (bf16*)carve((size_t)N_TOK * DETER * 2);
  float* yb     = (float*)carve((size_t)N_TOK * HID * 4);
  float* hraw   = (float*)carve((size_t)N_TOK * DETER * 4);
  float* gates  = (float*)carve((size_t)N_TOK * G_GRP * 3 * BH * 4);

  // ---- activations -> bf16 ----
  k_cvt_bf16<<<2048, 256, 0, stream>>>(s, s_bf, (size_t)N_TOK * 1024);
  k_cvt_bf16<<<2048, 256, 0, stream>>>(a, a_bf, (size_t)N_TOK * 256);
  k_cvt_bf16<<<2048, 256, 0, stream>>>(d, d_bf, (size_t)N_TOK * 4096);

  // ---- weights -> transposed bf16 (out, in) ----
  k_transpose_bf16<<<dim3(1024 / 32, 1024 / 32, 1), 256, 0, stream>>>(W_s, WsT, 1024, 1024);
  k_transpose_bf16<<<dim3(1024 / 32,  256 / 32, 1), 256, 0, stream>>>(W_a, WaT,  256, 1024);
  k_transpose_bf16<<<dim3(1024 / 32, 4096 / 32, 1), 256, 0, stream>>>(W_d, WdT, 4096, 1024);
  k_transpose_bf16<<<dim3(BH / 32, BLK_IN / 32, G_GRP), 256, 0, stream>>>(W_hid, WhidT, BLK_IN, BH);
  k_transpose_bf16<<<dim3(3 * BH / 32, BH / 32, G_GRP), 256, 0, stream>>>(W_gate, WgateT, BH, 3 * BH);

  // ---- input projections + LN + SiLU -> x_cat[:, {0,1024,2048}+1024) ----
  k_gemm_bf16<<<dim3(HID / 128, N_TOK / 128, 1), 256, 0, stream>>>(
      s_bf, 1024, 1024, 0, nullptr, 0, 0, 0, WsT, 1024, 0, yb, HID, 0);
  k_ln_silu<<<N_TOK, 256, 0, stream>>>(yb, HID, ln_s_g, ln_s_b, xcat, 3 * HID, 0);

  k_gemm_bf16<<<dim3(HID / 128, N_TOK / 128, 1), 256, 0, stream>>>(
      a_bf, 256, 256, 0, nullptr, 0, 0, 0, WaT, 256, 0, yb, HID, 0);
  k_ln_silu<<<N_TOK, 256, 0, stream>>>(yb, HID, ln_a_g, ln_a_b, xcat, 3 * HID, HID);

  k_gemm_bf16<<<dim3(HID / 128, N_TOK / 128, 1), 256, 0, stream>>>(
      d_bf, 4096, 4096, 0, nullptr, 0, 0, 0, WdT, 4096, 0, yb, HID, 0);
  k_ln_silu<<<N_TOK, 256, 0, stream>>>(yb, HID, ln_d_g, ln_d_b, xcat, 3 * HID, 2 * HID);

  // ---- block-diagonal hidden GEMM: A = [d_g (512) | x_cat (3072)] ----
  k_gemm_bf16<<<dim3(BH / 128, N_TOK / 128, G_GRP), 256, 0, stream>>>(
      d_bf, 4096, BH, BH,                 // A1: g-th 512-slice of d
      xcat, 3 * HID, 3 * HID, 0,          // A2: shared concat
      WhidT, BLK_IN, (long)BH * BLK_IN,   // BT per g
      hraw, DETER, BH);                   // C: g-th 512-slice of h
  k_ln_silu<<<N_TOK, 256, 0, stream>>>(hraw, DETER, ln_h_g, ln_h_b, hact, DETER, 0);

  // ---- block-diagonal gate GEMM ----
  k_gemm_bf16<<<dim3(3 * BH / 128, N_TOK / 128, G_GRP), 256, 0, stream>>>(
      hact, 4096, BH, BH,
      nullptr, 0, 0, 0,
      WgateT, BH, (long)3 * BH * BH,
      gates, G_GRP * 3 * BH, 3 * BH);

  // ---- GRU update epilogue ----
  k_gru_update<<<4096, 256, 0, stream>>>(gates, b_gate, d, out,
                                         (size_t)N_TOK * DETER);
}
